// FastSlowSurpriseKimiDeltaAttention_15848429322344
// MI455X (gfx1250) — matrix operand order repaired and checked by reading, not compiled, not run on hardware
//
#include <hip/hip_runtime.h>
#include <hip/hip_bf16.h>

// ---------------------------------------------------------------------------
// FastSlowSurpriseKimiDeltaAttention for MI455X (gfx1250, wave32)
//   - big GEMMs: 128x128 workgroup tiles, TDM (tensor_load_to_lds) double
//     buffering into LDS, v_wmma_f32_16x16x32_bf16 compute
//   - weights stored transposed (B^T) so fragments are contiguous 16B chunks
//   - sequential delta-rule scan: 1 workgroup per (b,h,speed), S in VGPRs
// ---------------------------------------------------------------------------

#define HN 16
#define DKN 128
#define DVN 128
#define BINSN 64
#define LDSPAD 40   // LDS row stride in bf16 elems: 32 data + 8 pad (TDM-inserted)

typedef __bf16 bf16;
typedef unsigned int u32;
typedef __attribute__((ext_vector_type(16))) __bf16 v16bf;
typedef __attribute__((ext_vector_type(8)))  __bf16 v8bf;
typedef __attribute__((ext_vector_type(8)))  float  v8f;
typedef __attribute__((ext_vector_type(4)))  u32    u32x4;
typedef __attribute__((ext_vector_type(8)))  int    i32x8;
typedef __attribute__((ext_vector_type(4)))  int    i32x4;

__device__ __forceinline__ float sigmoid_f(float x) { return 1.0f / (1.0f + expf(-x)); }
__device__ __forceinline__ float silu_f(float x)    { return x * sigmoid_f(x); }

__device__ __forceinline__ v16bf cat8(v8bf lo, v8bf hi) {
    return __builtin_shufflevector(lo, hi, 0, 1, 2, 3, 4, 5, 6, 7,
                                          8, 9, 10, 11, 12, 13, 14, 15);
}

// TDM: DMA a 32(K) x 128(rows) bf16 tile (row stride K elems) into LDS,
// padding each 64B row with 16B so LDS row stride is 80B.
// D# packed per CDNA5 ISA 8.3/8.4 (group0: count/lds/global/type,
// group1: data_size=2B, pad 16dw->+4dw, dims/strides).
// This toolchain exposes the 6-arg builtin:
//   (u32x4 g0, i32x8 g1, i32x4 g2, i32x4 g3, i32x8 extra, int cpol)
__device__ __forceinline__ void tdm_load_tile(const bf16* g, u32 lds_off, int K) {
    unsigned long long ga = (unsigned long long)(uintptr_t)g;
    u32x4 g0;
    g0[0] = 1u;                                            // count=1 (valid user D#)
    g0[1] = lds_off;                                       // lds_addr (bytes)
    g0[2] = (u32)ga;                                       // global_addr[31:0]
    g0[3] = (u32)((ga >> 32) & 0x01FFFFFFu) | (2u << 30);  // addr[56:32] | type=2
    i32x8 g1;
    g1[0] = (int)((1u << 16)      // data_size = 2 bytes
                | (1u << 20)      // pad_enable
                | (3u << 22)      // pad_interval: 16 DWORDs (64B)
                | (3u << 25));    // pad_amount: 4 DWORDs (16B)
    g1[1] = (int)(32u << 16);     // tensor_dim0 = 32 (lo16)
    g1[2] = (int)(128u << 16);    // tensor_dim0 hi | tensor_dim1 = 128 (lo16)
    g1[3] = (int)(32u << 16);     // tensor_dim1 hi | tile_dim0 = 32
    g1[4] = (int)128u;            // tile_dim1 = 128, tile_dim2 = 0
    g1[5] = (int)K;               // tensor_dim0_stride[31:0] = K elems
    g1[6] = 0;                    // stride hi / dim1_stride lo
    g1[7] = 0;
    i32x4 z4 = {0, 0, 0, 0};
    i32x8 z8 = {0, 0, 0, 0, 0, 0, 0, 0};
    __builtin_amdgcn_tensor_load_to_lds(g0, g1, z4, z4, z8, 0);
}

// ---------------------------------------------------------------- conversion
__global__ void k_f32_to_bf16(const float* __restrict__ in, bf16* __restrict__ out, int n) {
    int i = blockIdx.x * blockDim.x + threadIdx.x;
    if (i < n) out[i] = (bf16)in[i];
}

// Transpose-convert: in [K,N] f32 row-major -> out [N,K] bf16 row-major.
__global__ void k_cvt_transpose(const float* __restrict__ in, bf16* __restrict__ out,
                                int K, int N) {
    __shared__ float tile[16][17];
    const int kt = blockIdx.x * 16, nt = blockIdx.y * 16;
    const int tx = threadIdx.x, ty = threadIdx.y;
    tile[ty][tx] = in[(size_t)(kt + ty) * N + (nt + tx)];
    __syncthreads();
    out[(size_t)(nt + ty) * K + (kt + tx)] = (bf16)tile[tx][ty];
}

// --------------------------------------------------- TDM + LDS tiled WMMA GEMM
// C[M,N] = A[M,K] @ B[K,N] (+bias); A bf16 [M,K], Bt bf16 [N,K].
// Block: 256 thr (8 waves); C macro-tile 128x128; wave = 16(M) x 64(N) strip.
// Requires M%128==0, N%128==0, K%32==0.
__global__ void __launch_bounds__(256) k_wmma_gemm_lds(
        const bf16* __restrict__ A, const bf16* __restrict__ Bt,
        const float* __restrict__ bias, float* __restrict__ C,
        int M, int N, int K) {
    __shared__ __align__(16) bf16 Atile[2][128][LDSPAD];
    __shared__ __align__(16) bf16 Btile[2][128][LDSPAD];

    const int tid  = threadIdx.x;
    const int lane = tid & 31;
    const int wave = tid >> 5;          // 0..7
    const int half = lane >> 4;
    const int l16  = lane & 15;
    const int wm   = wave & 3;          // M sub-tile
    const int wn   = wave >> 2;         // N half
    const int Mblk = blockIdx.y * 128;
    const int Nblk = blockIdx.x * 128;
    const int nk = K / 32;

    if (tid == 0) {   // wave0 issues DMA; TDM ignores EXEC
        tdm_load_tile(A  + (size_t)Mblk * K, (u32)(uintptr_t)&Atile[0][0][0], K);
        tdm_load_tile(Bt + (size_t)Nblk * K, (u32)(uintptr_t)&Btile[0][0][0], K);
    }

    v8f acc[4] = {{}, {}, {}, {}};
    for (int i = 0; i < nk; ++i) {
        const int buf = i & 1;
        if (tid == 0) __builtin_amdgcn_s_wait_tensorcnt(0);
        __syncthreads();                               // tile[buf] ready for all waves
        if (tid == 0 && (i + 1) < nk) {                // prefetch next tile via TDM
            const int kt2 = (i + 1) * 32;
            tdm_load_tile(A  + (size_t)Mblk * K + kt2,
                          (u32)(uintptr_t)&Atile[buf ^ 1][0][0], K);
            tdm_load_tile(Bt + (size_t)Nblk * K + kt2,
                          (u32)(uintptr_t)&Btile[buf ^ 1][0][0], K);
        }
        const bf16* As = &Atile[buf][wm * 16 + l16][0];
        v16bf a = cat8(*(const v8bf*)(As + half * 8),
                       *(const v8bf*)(As + 16 + half * 8));
        #pragma unroll
        for (int j = 0; j < 4; ++j) {
            const bf16* Bs = &Btile[buf][wn * 64 + 16 * j + l16][0];
            v16bf b = cat8(*(const v8bf*)(Bs + half * 8),
                           *(const v8bf*)(Bs + 16 + half * 8));
            acc[j] = __builtin_amdgcn_wmma_f32_16x16x32_bf16(false, a, false, b,
                                                             (short)0, acc[j],
                                                             false, false);
        }
        __syncthreads();                               // done reading tile[buf]
    }
    #pragma unroll
    for (int j = 0; j < 4; ++j) {
        const int n = Nblk + wn * 64 + 16 * j + l16;
        const float bv = bias ? bias[n] : 0.0f;
        #pragma unroll
        for (int r = 0; r < 8; ++r)
            C[(size_t)(Mblk + wm * 16 + r + half * 8) * N + n] = acc[j][r] + bv;
    }
}

// -------------------------------------------- direct-global WMMA GEMM (small N)
// One wave per 16x64 strip. Used when N%128 != 0 (unc projection, N=64).
__global__ void k_wmma_gemm(const bf16* __restrict__ A, const bf16* __restrict__ Bt,
                            const float* __restrict__ bias, float* __restrict__ C,
                            int M, int N, int K) {
    const int lane = threadIdx.x;
    const int half = lane >> 4;
    const int l16  = lane & 15;
    const int mt = blockIdx.y * 16;
    const int nt = blockIdx.x * 64;

    const bf16* __restrict__ Arow = A + (size_t)(mt + l16) * K;
    const bf16* __restrict__ Brow[4];
    #pragma unroll
    for (int j = 0; j < 4; ++j)
        Brow[j] = Bt + (size_t)(nt + 16 * j + l16) * K;

    v8f acc[4] = {{}, {}, {}, {}};
    for (int kt = 0; kt < K; kt += 32) {
        if (kt + 32 < K) {
            __builtin_prefetch(Arow + kt + 32, 0, 1);
            __builtin_prefetch(Brow[0] + kt + 32, 0, 1);
            __builtin_prefetch(Brow[2] + kt + 32, 0, 1);
        }
        v16bf a = cat8(*(const v8bf*)(Arow + kt + half * 8),
                       *(const v8bf*)(Arow + kt + 16 + half * 8));
        #pragma unroll
        for (int j = 0; j < 4; ++j) {
            const bf16* bp = Brow[j] + kt + half * 8;
            v16bf b = cat8(*(const v8bf*)bp, *(const v8bf*)(bp + 16));
            acc[j] = __builtin_amdgcn_wmma_f32_16x16x32_bf16(false, a, false, b,
                                                             (short)0, acc[j],
                                                             false, false);
        }
    }
    #pragma unroll
    for (int j = 0; j < 4; ++j) {
        const int n = nt + 16 * j + l16;
        const float bv = bias ? bias[n] : 0.0f;
        #pragma unroll
        for (int r = 0; r < 8; ++r)
            C[(size_t)(mt + r + half * 8) * N + n] = acc[j][r] + bv;
    }
}

// --------------------------------------------- depthwise causal conv (K=4) + SiLU
__global__ void k_conv_silu(const float* __restrict__ pre, const float* __restrict__ w,
                            float* __restrict__ out, int Bn, int T, int C) {
    int i = blockIdx.x * blockDim.x + threadIdx.x;
    int total = Bn * T * C;
    if (i >= total) return;
    int c = i % C;
    int t = (i / C) % T;
    int b = i / (C * T);
    float acc = 0.0f;
    #pragma unroll
    for (int j = 0; j < 4; ++j) {
        int tt = t - 3 + j;
        float xv = (tt >= 0) ? pre[((size_t)b * T + tt) * C + c] : 0.0f;
        acc = fmaf(xv, w[c * 4 + j], acc);
    }
    out[i] = silu_f(acc);
}

// ------------------------------------------------- entropy of 64-bin softmax
__global__ void k_entropy(const float* __restrict__ logits, float* __restrict__ entn, int rows) {
    int r = blockIdx.x * blockDim.x + threadIdx.x;
    if (r >= rows) return;
    const float* p = logits + (size_t)r * BINSN;
    float m = -3.4e38f;
    for (int i = 0; i < BINSN; ++i) m = fmaxf(m, p[i]);
    float s = 0.0f;
    for (int i = 0; i < BINSN; ++i) s += expf(p[i] - m);
    float lse = m + logf(s);
    float e = 0.0f;
    for (int i = 0; i < BINSN; ++i) { float lp = p[i] - lse; e -= expf(lp) * lp; }
    entn[r] = e / logf((float)BINSN);
}

// ------------------- per-(b,t,h): norms, k_unit, feats -> MLP -> beta/lam/decay
__global__ void k_features(const float* __restrict__ kpost, const float* __restrict__ vpost,
                           const float* __restrict__ vhat, const float* __restrict__ entn,
                           const float* __restrict__ head_embed,
                           const float* __restrict__ w1, const float* __restrict__ b1,
                           const float* __restrict__ w2, const float* __restrict__ b2,
                           const float* __restrict__ bbw, const float* __restrict__ bbb,
                           const float* __restrict__ bdw, const float* __restrict__ bdb,
                           const float* __restrict__ lamw, const float* __restrict__ lamb,
                           const float* __restrict__ abw, const float* __restrict__ abb,
                           const float* __restrict__ adw, const float* __restrict__ adb,
                           const float* __restrict__ A_log, const float* __restrict__ dt_bias,
                           float* __restrict__ kunit, float* __restrict__ betaf,
                           float* __restrict__ betas, float* __restrict__ lamo,
                           float* __restrict__ decf, float* __restrict__ decs, int BT) {
    int i = blockIdx.x * blockDim.x + threadIdx.x;   // over BT*H
    if (i >= BT * HN) return;
    int h = i % HN;
    int bt = i / HN;
    size_t base = (size_t)i * DKN;

    float kn = 0, vn = 0, vhn = 0, dot = 0, en = 0;
    for (int m = 0; m < DKN; ++m) {
        float kv = kpost[base + m], vv = vpost[base + m], vh = vhat[base + m];
        kn = fmaf(kv, kv, kn); vn = fmaf(vv, vv, vn); vhn = fmaf(vh, vh, vhn);
        dot = fmaf(vv, vh, dot);
        float e = vv - vh; en = fmaf(e, e, en);
    }
    kn = sqrtf(kn); vn = sqrtf(vn); vhn = sqrtf(vhn); en = sqrtf(en);
    float inv = 1.0f / (kn + 1e-6f);
    for (int m = 0; m < DKN; ++m) kunit[base + m] = kpost[base + m] * inv;

    float feats[8];
    feats[0] = entn[bt];
    feats[1] = en / (vn + 1e-6f);
    feats[2] = dot / (vn * vhn + 1e-6f);
    feats[3] = log1pf(en);
    #pragma unroll
    for (int j = 0; j < 4; ++j) feats[4 + j] = head_embed[h * 4 + j];

    float h1[32];
    #pragma unroll 4
    for (int o = 0; o < 32; ++o) {
        float a = b1[o];
        #pragma unroll
        for (int j = 0; j < 8; ++j) a = fmaf(feats[j], w1[j * 32 + o], a);
        h1[o] = silu_f(a);
    }
    float h2[32];
    #pragma unroll 2
    for (int o = 0; o < 32; ++o) {
        float a = b2[o];
        #pragma unroll 8
        for (int j = 0; j < 32; ++j) a = fmaf(h1[j], w2[j * 32 + o], a);
        h2[o] = silu_f(a);
    }
    float bb = bbb[0], bd = bdb[0], la = lamb[0], ab = abb[0], ad = adb[0];
    #pragma unroll 8
    for (int j = 0; j < 32; ++j) {
        bb = fmaf(h2[j], bbw[j], bb);
        bd = fmaf(h2[j], bdw[j], bd);
        la = fmaf(h2[j], lamw[j], la);
        ab = fmaf(h2[j], abw[j], ab);
        ad = fmaf(h2[j], adw[j], ad);
    }
    float bfst = sigmoid_f(bb + bd), bslw = sigmoid_f(bb - bd);
    float afst = sigmoid_f(ab + ad), aslw = sigmoid_f(ab - ad);
    betaf[i] = bfst; betas[i] = bslw; lamo[i] = sigmoid_f(la);

    float negA = -expf(A_log[h]);
    for (int m = 0; m < DKN; ++m) {
        float sp = log1pf(expf(dt_bias[h * DKN + m]));  // softplus
        float gch = negA * sp;
        decf[base + m] = expf(gch * afst);
        decs[base + m] = expf(gch * aslw);
    }
}

// ---------------------------------------------------- sequential delta rule
__global__ void __launch_bounds__(128) k_delta(
        const float* __restrict__ q, const float* __restrict__ kunit,
        const float* __restrict__ v,
        const float* __restrict__ betaf, const float* __restrict__ betas,
        const float* __restrict__ decf,  const float* __restrict__ decs,
        float* __restrict__ ofast, float* __restrict__ oslow, int Bn, int T) {
    const int spd = blockIdx.x & 1;
    const int bh  = blockIdx.x >> 1;
    const int b = bh / HN, h = bh % HN;
    const float* __restrict__ beta = spd ? betas : betaf;
    const float* __restrict__ dec  = spd ? decs  : decf;
    float* __restrict__ out        = spd ? oslow : ofast;
    const int vi = threadIdx.x;

    __shared__ float skk[DKN], sqq[DKN], sdd[DKN];
    float S[DKN];
    #pragma unroll
    for (int k = 0; k < DKN; ++k) S[k] = 0.0f;
    const float qscale = 0.08838834764831845f;  // 128^-0.5

    for (int t = 0; t < T; ++t) {
        const size_t base = ((size_t)(b * T + t) * HN + h) * (size_t)DKN;
        skk[vi] = kunit[base + vi];
        sqq[vi] = q[base + vi] * qscale;
        sdd[vi] = dec[base + vi];
        __syncthreads();
        const float vt  = v[base + vi];
        const float bt_ = beta[(size_t)(b * T + t) * HN + h];
        float vp = 0.0f;
        #pragma unroll
        for (int k = 0; k < DKN; ++k) { S[k] *= sdd[k]; vp = fmaf(skk[k], S[k], vp); }
        const float dvb = bt_ * (vt - vp);
        float o = 0.0f;
        #pragma unroll
        for (int k = 0; k < DKN; ++k) { S[k] = fmaf(skk[k], dvb, S[k]); o = fmaf(sqq[k], S[k], o); }
        out[base + vi] = o;
        __syncthreads();
    }
}

// ------------------------------------- mix fast/slow, RMS norm, gate, -> bf16
__global__ void k_gate(const float* __restrict__ ofast, const float* __restrict__ oslow,
                       const float* __restrict__ lam, const float* __restrict__ g,
                       const float* __restrict__ onw, bf16* __restrict__ ofin, int BT) {
    int i = blockIdx.x * blockDim.x + threadIdx.x;   // over BT*H
    if (i >= BT * HN) return;
    size_t base = (size_t)i * DVN;                   // == bt*2048 + h*128
    float l = lam[i];
    float ss = 0.0f;
    for (int m = 0; m < DVN; ++m) {
        float o = fmaf(l, ofast[base + m] - oslow[base + m], oslow[base + m]);
        ss = fmaf(o, o, ss);
    }
    float rinv = rsqrtf(ss * (1.0f / DVN) + 1e-5f);
    for (int m = 0; m < DVN; ++m) {
        float o = fmaf(l, ofast[base + m] - oslow[base + m], oslow[base + m]);
        float r = o * rinv * onw[m] * sigmoid_f(g[base + m]);
        ofin[base + m] = (bf16)r;
    }
}

// ---------------------------------------------------------------------------
extern "C" void kernel_launch(void* const* d_in, const int* in_sizes, int n_in,
                              void* d_out, int out_size, void* d_ws, size_t ws_size,
                              hipStream_t stream) {
    const int Bn = 2, T = 256, D = 2048;
    const int BT = Bn * T;                 // 512 rows

    const float* x        = (const float*)d_in[0];
    const float* wq       = (const float*)d_in[1];
    const float* wk       = (const float*)d_in[2];
    const float* wv       = (const float*)d_in[3];
    const float* conv_q_w = (const float*)d_in[4];
    const float* conv_k_w = (const float*)d_in[5];
    const float* conv_v_w = (const float*)d_in[6];
    const float* A_log    = (const float*)d_in[7];
    const float* dt_bias  = (const float*)d_in[8];
    const float* proxy_w  = (const float*)d_in[9];
    const float* unc_w    = (const float*)d_in[10];
    const float* head_emb = (const float*)d_in[11];
    const float* mlp_w1   = (const float*)d_in[12];
    const float* mlp_b1   = (const float*)d_in[13];
    const float* mlp_w2   = (const float*)d_in[14];
    const float* mlp_b2   = (const float*)d_in[15];
    const float* bb_w     = (const float*)d_in[16];
    const float* bb_b     = (const float*)d_in[17];
    const float* bd_w     = (const float*)d_in[18];
    const float* bd_b     = (const float*)d_in[19];
    const float* lam_w    = (const float*)d_in[20];
    const float* lam_b    = (const float*)d_in[21];
    const float* ab_w     = (const float*)d_in[22];
    const float* ab_b     = (const float*)d_in[23];
    const float* ad_w     = (const float*)d_in[24];
    const float* ad_b     = (const float*)d_in[25];
    const float* g1_w     = (const float*)d_in[26];
    const float* g2_w     = (const float*)d_in[27];
    const float* g2_b     = (const float*)d_in[28];
    const float* onorm_w  = (const float*)d_in[29];
    const float* wo       = (const float*)d_in[30];

    char* wsp = (char*)d_ws;
    auto alloc = [&](size_t bytes) -> void* {
        void* p = (void*)wsp;
        wsp += (bytes + 255) & ~(size_t)255;
        return p;
    };
    bf16* xb    = (bf16*)alloc((size_t)BT * D * 2);       // [BT,D]
    bf16* wqbT  = (bf16*)alloc((size_t)D * D * 2);        // [N,K] transposed
    bf16* wkbT  = (bf16*)alloc((size_t)D * D * 2);
    bf16* wvbT  = (bf16*)alloc((size_t)D * D * 2);
    bf16* wobT  = (bf16*)alloc((size_t)D * D * 2);
    bf16* uncwbT = (bf16*)alloc((size_t)BINSN * D * 2);
    bf16* g1wbT  = (bf16*)alloc((size_t)DVN * D * 2);
    bf16* g2wbT  = (bf16*)alloc((size_t)D * DVN * 2);
    bf16* pwbT   = (bf16*)alloc((size_t)DVN * DKN * 2);
    float* qpre = (float*)alloc((size_t)BT * D * 4);
    float* kpre = (float*)alloc((size_t)BT * D * 4);
    float* vpre = (float*)alloc((size_t)BT * D * 4);
    float* qpost = (float*)alloc((size_t)BT * D * 4);
    float* kpost = (float*)alloc((size_t)BT * D * 4);
    float* vpost = (float*)alloc((size_t)BT * D * 4);
    float* unc  = (float*)alloc((size_t)BT * BINSN * 4);
    float* entn = (float*)alloc((size_t)BT * 4);
    float* g1o  = (float*)alloc((size_t)BT * DVN * 4);
    bf16*  g1b  = (bf16*)alloc((size_t)BT * DVN * 2);
    float* garr = (float*)alloc((size_t)BT * D * 4);
    bf16*  kb   = (bf16*)alloc((size_t)BT * D * 2);
    float* vhat = (float*)alloc((size_t)BT * D * 4);
    float* kunit = (float*)alloc((size_t)BT * D * 4);
    float* betaf = (float*)alloc((size_t)BT * HN * 4);
    float* betas = (float*)alloc((size_t)BT * HN * 4);
    float* lamv  = (float*)alloc((size_t)BT * HN * 4);
    float* decf  = (float*)alloc((size_t)BT * D * 4);
    float* decs  = (float*)alloc((size_t)BT * D * 4);
    float* ofastA = (float*)alloc((size_t)BT * D * 4);
    float* oslowA = (float*)alloc((size_t)BT * D * 4);
    bf16*  ofinb  = (bf16*)alloc((size_t)BT * D * 2);

    auto cvt = [&](const float* src, bf16* dst, int n) {
        k_f32_to_bf16<<<(n + 255) / 256, 256, 0, stream>>>(src, dst, n);
    };
    auto cvtT = [&](const float* src, bf16* dst, int K, int N) {   // [K,N] -> [N,K]
        dim3 grid(K / 16, N / 16);
        k_cvt_transpose<<<grid, dim3(16, 16), 0, stream>>>(src, dst, K, N);
    };
    auto gemm_lds = [&](const bf16* A, const bf16* BtM, const float* bias, float* C,
                        int M, int N, int K) {      // M%128==0, N%128==0, K%32==0
        dim3 grid(N / 128, M / 128);
        k_wmma_gemm_lds<<<grid, 256, 0, stream>>>(A, BtM, bias, C, M, N, K);
    };
    auto gemm_dir = [&](const bf16* A, const bf16* BtM, const float* bias, float* C,
                        int M, int N, int K) {      // M%16==0, N%64==0, K%32==0
        dim3 grid(N / 64, M / 16);
        k_wmma_gemm<<<grid, 32, 0, stream>>>(A, BtM, bias, C, M, N, K);
    };

    // 1) bf16 conversions: activations row-major, weights transposed
    cvt(x, xb, BT * D);
    cvtT(wq, wqbT, D, D);
    cvtT(wk, wkbT, D, D);
    cvtT(wv, wvbT, D, D);
    cvtT(wo, wobT, D, D);
    cvtT(unc_w, uncwbT, D, BINSN);
    cvtT(g1_w, g1wbT, D, DVN);
    cvtT(g2_w, g2wbT, DVN, D);
    cvtT(proxy_w, pwbT, DKN, DVN);

    // 2) projection GEMMs (TDM + WMMA)
    gemm_lds(xb, wqbT, nullptr, qpre, BT, D, D);
    gemm_lds(xb, wkbT, nullptr, kpre, BT, D, D);
    gemm_lds(xb, wvbT, nullptr, vpre, BT, D, D);
    gemm_dir(xb, uncwbT, nullptr, unc, BT, BINSN, D);   // N=64 -> direct path
    gemm_lds(xb, g1wbT, nullptr, g1o, BT, DVN, D);

    // 3) depthwise causal conv + SiLU
    int nconv = BT * D;
    k_conv_silu<<<(nconv + 255) / 256, 256, 0, stream>>>(qpre, conv_q_w, qpost, Bn, T, D);
    k_conv_silu<<<(nconv + 255) / 256, 256, 0, stream>>>(kpre, conv_k_w, kpost, Bn, T, D);
    k_conv_silu<<<(nconv + 255) / 256, 256, 0, stream>>>(vpre, conv_v_w, vpost, Bn, T, D);

    // 4) uncertainty entropy
    k_entropy<<<(BT + 255) / 256, 256, 0, stream>>>(unc, entn, BT);

    // 5) gate path: g = (x@g1_w)@g2_w + g2_b
    cvt(g1o, g1b, BT * DVN);
    gemm_lds(g1b, g2wbT, g2_b, garr, BT, D, DVN);

    // 6) v_hat = k @ proxy_w  (rows = B*T*H, per-head contiguous 128)
    cvt(kpost, kb, BT * D);
    gemm_lds(kb, pwbT, nullptr, vhat, BT * HN, DVN, DKN);

    // 7) features -> MLP -> beta/lam/amp/decay + k_unit
    int nbth = BT * HN;
    k_features<<<(nbth + 255) / 256, 256, 0, stream>>>(
        kpost, vpost, vhat, entn, head_emb,
        mlp_w1, mlp_b1, mlp_w2, mlp_b2,
        bb_w, bb_b, bd_w, bd_b, lam_w, lam_b, ab_w, ab_b, ad_w, ad_b,
        A_log, dt_bias,
        kunit, betaf, betas, lamv, decf, decs, BT);

    // 8) sequential delta rule (fast + slow interleaved across blocks)
    k_delta<<<Bn * HN * 2, 128, 0, stream>>>(qpost, kunit, vpost, betaf, betas,
                                             decf, decs, ofastA, oslowA, Bn, T);

    // 9) mix, RMS-norm, gate -> bf16
    k_gate<<<(nbth + 255) / 256, 256, 0, stream>>>(ofastA, oslowA, lamv, garr,
                                                   onorm_w, ofinb, BT);

    // 10) output projection
    gemm_lds(ofinb, wobT, nullptr, (float*)d_out, BT, D, D);
}
